// RelPositionMultiHeadedAttention_54915451847150
// MI455X (gfx1250) — compile-verified, matrix operand order checked by reading, not run on hardware
//
#include <hip/hip_runtime.h>
#include <hip/hip_bf16.h>

// RelPositionMultiHeadedAttention for MI455X (gfx1250), wave32 + WMMA + async LDS.
// B=2, T=2048, E=512, H=8, D=64, PAD=128.
//
// Pipeline:
//   1) proj_gemm (x4): fp32 GEMM inputs -> f16 activations via v_wmma_f32_16x16x32_f16
//        q -> qu=[q+pos_bias_u], qv=[q+pos_bias_v]   (layout [B,H,T,D] f16)
//        k -> [B,H,T,D] f16 ;  p -> [B,H,T,D] f16 ;  v -> transposed [B,H,D,T] f16
//   2) relattn: flash-attention with Transformer-XL relative shift
//        - rolling 2-tile LDS window for the sheared rel-pos raw scores
//        - V^T blocks staged to LDS with global_load_async_to_lds_b128 (double buffered,
//          synchronized with s_wait_asynccnt)
//   3) out_gemm: f16 attention output x w_out^T + b_out -> fp32 d_out

#define B_   2
#define T_   2048
#define E_   512
#define H_   8
#define D_   64
#define PAD_ 128

typedef __attribute__((ext_vector_type(16))) _Float16 v16h;
typedef __attribute__((ext_vector_type(8)))  float    v8f;

static __device__ inline v8f wmma_f16(v16h a, v16h b, v8f c) {
  // D = A(16x32 f16) * B(32x16 f16) + C(16x16 f32)
  return __builtin_amdgcn_wmma_f32_16x16x32_f16(false, a, false, b, (short)0, c,
                                                false, false);
}

// ---- fragment loaders ------------------------------------------------------
// A / B fragment (16x32 f16). Per ISA: lane L -> row M = L&15; K chunks
// [base0, base0+8) and [base0+16, base0+24) with base0 = (L<16) ? 0 : 8.

static __device__ inline v16h load_frag_f32(const float* __restrict__ base,
                                            int row0, int k0, int ld) {
  const int lane = threadIdx.x & 31;
  const int m    = lane & 15;
  const int b0   = (lane < 16) ? 0 : 8;
  const float* p = base + (long)(row0 + m) * ld + k0 + b0;
  v16h f;
#pragma unroll
  for (int i = 0; i < 8; ++i) f[i] = (_Float16)p[i];
#pragma unroll
  for (int i = 0; i < 8; ++i) f[i + 8] = (_Float16)p[i + 16];
  return f;
}

static __device__ inline v16h load_frag_f16(const _Float16* __restrict__ base,
                                            int row0, int k0, int ld) {
  const int lane = threadIdx.x & 31;
  const int m    = lane & 15;
  const int b0   = (lane < 16) ? 0 : 8;
  const _Float16* p = base + (long)(row0 + m) * ld + k0 + b0;
  v16h f;
#pragma unroll
  for (int i = 0; i < 8; ++i) f[i] = p[i];
#pragma unroll
  for (int i = 0; i < 8; ++i) f[i + 8] = p[i + 16];
  return f;
}

static __device__ inline v16h load_frag_f16_clamp(const _Float16* __restrict__ base,
                                                  int row0, int k0, int ld,
                                                  int maxrow) {
  const int lane = threadIdx.x & 31;
  const int m    = lane & 15;
  const int row  = row0 + m;
  const bool ok  = (row >= 0) && (row < maxrow);
  const _Float16* p = base + (long)(ok ? row : 0) * ld + k0 + ((lane < 16) ? 0 : 8);
  v16h f;
#pragma unroll
  for (int i = 0; i < 8; ++i) f[i] = ok ? p[i] : (_Float16)0.f;
#pragma unroll
  for (int i = 0; i < 8; ++i) f[i + 8] = ok ? p[i + 16] : (_Float16)0.f;
  return f;
}

// 16-lane group reductions (wave32: xor masks 1,2,4,8 stay within each half).
static __device__ inline float rmax16(float v) {
#pragma unroll
  for (int m = 1; m < 16; m <<= 1) v = fmaxf(v, __shfl_xor(v, m, 32));
  return v;
}
static __device__ inline float rsum16(float v) {
#pragma unroll
  for (int m = 1; m < 16; m <<= 1) v += __shfl_xor(v, m, 32);
  return v;
}

// ---- async V^T block staging: global [64 rows x 32 cols] f16 -> LDS --------
// 256 chunks of 16B; 32 lanes x 8 async instructions. Tracked by ASYNCcnt.
static __device__ inline void issue_vstage(const _Float16* __restrict__ vp,
                                           int j0, _Float16* ldsbase) {
  const int lane = threadIdx.x & 31;
#pragma unroll
  for (int i = 0; i < 8; ++i) {
    const int chunk = i * 32 + lane;
    const int row   = chunk >> 2;        // d in [0,64)
    const int cc    = chunk & 3;         // 8-half column chunk
    const _Float16* g = vp + (long)row * T_ + j0 + cc * 8;
    // Generic __shared__ address low 32 bits == wave-relative LDS byte offset
    // (ISA 10.2: the LDS aperture uses the low 32 address bits).
    const unsigned lds_off =
        (unsigned)(unsigned long)(ldsbase + row * 32 + cc * 8);
    asm volatile("global_load_async_to_lds_b128 %0, %1, off"
                 :: "v"(lds_off), "v"((unsigned long)g)
                 : "memory");
  }
}

// ---- projection GEMM: C[m,n] = A[m,:] . W[n,:] + bias[n] -------------------
// mode 0: out0[b,h,t,d] ; mode 1 (values): out0[b,h,d,t] (transposed) ;
// mode 3 (query): out0 = C + pos_bias_u, out1 = C + pos_bias_v, layout [b,h,t,d].
__global__ __launch_bounds__(128) void proj_gemm_kernel(
    const float* __restrict__ A, const float* __restrict__ W,
    const float* __restrict__ bias, const float* __restrict__ bu,
    const float* __restrict__ bv, _Float16* __restrict__ out0,
    _Float16* __restrict__ out1, int mode) {
  const int lane = threadIdx.x & 31;
  const int wv   = threadIdx.x >> 5;
  const int row0 = blockIdx.x * 64 + (wv & 1) * 32;
  const int col0 = blockIdx.y * 64 + (wv >> 1) * 32;

  v8f acc[2][2] = {};
  for (int k0 = 0; k0 < E_; k0 += 32) {
    v16h a0 = load_frag_f32(A, row0,      k0, E_);
    v16h a1 = load_frag_f32(A, row0 + 16, k0, E_);
    v16h b0 = load_frag_f32(W, col0,      k0, E_);
    v16h b1 = load_frag_f32(W, col0 + 16, k0, E_);
    acc[0][0] = wmma_f16(a0, b0, acc[0][0]);
    acc[0][1] = wmma_f16(a0, b1, acc[0][1]);
    acc[1][0] = wmma_f16(a1, b0, acc[1][0]);
    acc[1][1] = wmma_f16(a1, b1, acc[1][1]);
  }

  const int n   = lane & 15;
  const int hi8 = (lane < 16) ? 0 : 8;
#pragma unroll
  for (int i = 0; i < 2; ++i)
#pragma unroll
    for (int j = 0; j < 2; ++j) {
      const int colb  = col0 + j * 16 + n;
      const float bval = bias ? bias[colb] : 0.f;
#pragma unroll
      for (int r = 0; r < 8; ++r) {
        const int m  = row0 + i * 16 + r + hi8;
        const int bb = m >> 11;          // m / T_
        const int t  = m & (T_ - 1);
        const int h  = colb >> 6;        // colb / D_
        const int d  = colb & (D_ - 1);
        const float v = acc[i][j][r] + bval;
        if (mode == 3) {
          const long idx = (((long)(bb * H_ + h)) * T_ + t) * D_ + d;
          out0[idx] = (_Float16)(v + bu[colb]);
          out1[idx] = (_Float16)(v + bv[colb]);
        } else if (mode == 1) {
          out0[(((long)(bb * H_ + h)) * D_ + d) * T_ + t] = (_Float16)v;
        } else {
          out0[(((long)(bb * H_ + h)) * T_ + t) * D_ + d] = (_Float16)v;
        }
      }
    }
}

// ---- fused rel-pos flash attention ----------------------------------------
// One wave per 16-row query tile; streams 32 key columns per iteration.
__global__ __launch_bounds__(128) void relattn_kernel(
    const _Float16* __restrict__ qu, const _Float16* __restrict__ qv,
    const _Float16* __restrict__ kk, const _Float16* __restrict__ pp,
    const _Float16* __restrict__ vT, const unsigned char* __restrict__ mask,
    _Float16* __restrict__ attnO) {
  __shared__ __align__(16) float    rawLDS[4][16][32];     // rolling raw window
  __shared__ __align__(16) _Float16 pLDS[4][16][32];       // probs restage
  __shared__ __align__(16) _Float16 vstage[4][2][64][32];  // async V^T blocks

  const int lane = threadIdx.x & 31;
  const int wv   = threadIdx.x >> 5;
  const int tile = blockIdx.x * 4 + wv;     // [0, B*H*T/16)
  const int bh   = tile >> 7;               // 128 q-tiles per (b,h)
  const int q0   = (tile & 127) << 4;
  const int b    = bh >> 3;
  const int h    = bh & 7;

  const long baseBH = (long)bh * T_ * D_;
  const _Float16* qu_p = qu + baseBH;
  const _Float16* qv_p = qv + baseBH;
  const _Float16* k_p  = kk + baseBH;
  const _Float16* p_p  = pp + baseBH;
  const _Float16* v_p  = vT + baseBH;       // [D_, T_]
  const unsigned char* m_p = mask + (long)b * T_;

  const int n    = lane & 15;
  const int hi8  = (lane < 16) ? 0 : 8;
  const int offk = T_ - 16 - q0;            // 16-aligned shear offset
  const int wt0  = offk >> 4;               // absolute tile index of window base

  // resident query fragments (K-dim = D = 64 -> two 16x32 A-frags each)
  const v16h au0 = load_frag_f16(qu_p, q0, 0,  D_);
  const v16h au1 = load_frag_f16(qu_p, q0, 32, D_);
  const v16h av0 = load_frag_f16(qv_p, q0, 0,  D_);
  const v16h av1 = load_frag_f16(qv_p, q0, 32, D_);

  // prologue: stage first V^T block; compute first raw rel-pos tile (cols offk..)
  issue_vstage(v_p, 0, &vstage[wv][0][0][0]);
  {
    v8f raw = {};
    raw = wmma_f16(av0, load_frag_f16_clamp(p_p, offk, 0,  D_, T_), raw);
    raw = wmma_f16(av1, load_frag_f16_clamp(p_p, offk, 32, D_, T_), raw);
    const int slot = (wt0 & 1) << 4;
#pragma unroll
    for (int r = 0; r < 8; ++r) rawLDS[wv][r + hi8][slot + n] = raw[r];
  }

  v8f O[4] = {};
  float mrow[8], lrow[8];
#pragma unroll
  for (int r = 0; r < 8; ++r) { mrow[r] = -3.0e38f; lrow[r] = 0.f; }

  int buf = 0;
  for (int j0 = 0; j0 < T_; j0 += 32) {
    if (j0 + 32 < T_)
      issue_vstage(v_p, j0 + 32, &vstage[wv][buf ^ 1][0][0]);
    __builtin_prefetch(k_p + (long)(j0 + 32) * D_, 0, 1);

    v8f S[2];
#pragma unroll
    for (int half = 0; half < 2; ++half) {
      const int jj = j0 + half * 16;
      const int wt = wt0 + (jj >> 4);       // abs index of window's first tile
      // content scores: (q+u) . k^T
      v8f s = {};
      s = wmma_f16(au0, load_frag_f16(k_p, jj, 0,  D_), s);
      s = wmma_f16(au1, load_frag_f16(k_p, jj, 32, D_), s);
      // rolling raw window: compute only the NEW tile (abs index wt+1)
      {
        const int cb = jj + offk + 16;
        v8f raw = {};
        raw = wmma_f16(av0, load_frag_f16_clamp(p_p, cb, 0,  D_, T_), raw);
        raw = wmma_f16(av1, load_frag_f16_clamp(p_p, cb, 32, D_, T_), raw);
        const int slot = ((wt + 1) & 1) << 4;
#pragma unroll
        for (int r = 0; r < 8; ++r) rawLDS[wv][r + hi8][slot + n] = raw[r];
      }
      const float mb = m_p[jj + n] ? -1.0e30f : 0.f;
#pragma unroll
      for (int r = 0; r < 8; ++r) {
        const int M = r + hi8;
        // rel-shift gather: pos[a,c] = raw[a, c-a+T-1]; window col = n+15-M
        const int wcol = n + 15 - M;
        const int slot = ((wt + (wcol >> 4)) & 1) << 4;
        const float pos = ((jj + n) <= (q0 + M))
                              ? rawLDS[wv][M][slot + (wcol & 15)] : 0.f;
        s[r] = (s[r] + pos) * 0.125f + mb;   // 1/sqrt(D)=0.125
      }
      S[half] = s;
    }

    // online softmax over the 32 new columns
#pragma unroll
    for (int r = 0; r < 8; ++r) {
      const int M = r + hi8;
      const float tmax = fmaxf(rmax16(S[0][r]), rmax16(S[1][r]));
      const float mnew = fmaxf(mrow[r], tmax);
      const float scal = __expf(mrow[r] - mnew);
      const float p0   = __expf(S[0][r] - mnew);
      const float p1   = __expf(S[1][r] - mnew);
      lrow[r] = lrow[r] * scal + rsum16(p0) + rsum16(p1);
      mrow[r] = mnew;
      O[0][r] = O[0][r] * scal;
      O[1][r] = O[1][r] * scal;
      O[2][r] = O[2][r] * scal;
      O[3][r] = O[3][r] * scal;
      pLDS[wv][M][n]      = (_Float16)p0;
      pLDS[wv][M][n + 16] = (_Float16)p1;
    }

    // restage probs into A-frag layout
    v16h pf;
    {
      const int b0 = hi8;
#pragma unroll
      for (int i = 0; i < 8; ++i) {
        pf[i]     = pLDS[wv][n][b0 + i];
        pf[i + 8] = pLDS[wv][n][b0 + 16 + i];
      }
    }

    // wait for current V^T block (next block may still be in flight: cnt<=8)
    if (j0 + 32 < T_) asm volatile("s_wait_asynccnt 0x8" ::: "memory");
    else              asm volatile("s_wait_asynccnt 0x0" ::: "memory");

    const _Float16* vb = &vstage[wv][buf][0][0];
#pragma unroll
    for (int n0 = 0; n0 < 4; ++n0) {
      const int vrow = n0 * 16 + n;
      const _Float16* vp = vb + vrow * 32 + hi8;
      v16h vf;
#pragma unroll
      for (int i = 0; i < 8; ++i) { vf[i] = vp[i]; vf[i + 8] = vp[i + 16]; }
      O[n0] = wmma_f16(pf, vf, O[n0]);
    }
    buf ^= 1;
  }

  // epilogue: normalize and store f16 [B*T, E]
#pragma unroll
  for (int r = 0; r < 8; ++r) {
    const float inv = 1.f / lrow[r];
    const int a = q0 + r + hi8;
    const long rowb = ((long)b * T_ + a) * E_ + h * D_;
#pragma unroll
    for (int n0 = 0; n0 < 4; ++n0)
      attnO[rowb + n0 * 16 + n] = (_Float16)(O[n0][r] * inv);
  }
}

// ---- output GEMM: out[m,n] = AO[m,:] . Wout[n,:] + b_out[n] (fp32 out) -----
__global__ __launch_bounds__(128) void out_gemm_kernel(
    const _Float16* __restrict__ A, const float* __restrict__ W,
    const float* __restrict__ bias, float* __restrict__ out) {
  const int lane = threadIdx.x & 31;
  const int wv   = threadIdx.x >> 5;
  const int row0 = blockIdx.x * 64 + (wv & 1) * 32;
  const int col0 = blockIdx.y * 64 + (wv >> 1) * 32;

  v8f acc[2][2] = {};
  for (int k0 = 0; k0 < E_; k0 += 32) {
    v16h a0 = load_frag_f16(A, row0,      k0, E_);
    v16h a1 = load_frag_f16(A, row0 + 16, k0, E_);
    v16h b0 = load_frag_f32(W, col0,      k0, E_);
    v16h b1 = load_frag_f32(W, col0 + 16, k0, E_);
    acc[0][0] = wmma_f16(a0, b0, acc[0][0]);
    acc[0][1] = wmma_f16(a0, b1, acc[0][1]);
    acc[1][0] = wmma_f16(a1, b0, acc[1][0]);
    acc[1][1] = wmma_f16(a1, b1, acc[1][1]);
  }

  const int n   = lane & 15;
  const int hi8 = (lane < 16) ? 0 : 8;
#pragma unroll
  for (int i = 0; i < 2; ++i)
#pragma unroll
    for (int j = 0; j < 2; ++j) {
      const int colb = col0 + j * 16 + n;
      const float bval = bias[colb];
#pragma unroll
      for (int r = 0; r < 8; ++r) {
        const int m = row0 + i * 16 + r + hi8;
        out[(long)m * E_ + colb] = acc[i][j][r] + bval;
      }
    }
}

extern "C" void kernel_launch(void* const* d_in, const int* in_sizes, int n_in,
                              void* d_out, int out_size, void* d_ws, size_t ws_size,
                              hipStream_t stream) {
  (void)in_sizes; (void)n_in; (void)out_size; (void)ws_size;
  const float* query = (const float*)d_in[0];
  const float* key   = (const float*)d_in[1];
  const float* value = (const float*)d_in[2];
  const float* pos   = (const float*)d_in[3];
  const unsigned char* mask = (const unsigned char*)d_in[4];
  const float* w_q   = (const float*)d_in[5];
  const float* b_q   = (const float*)d_in[6];
  const float* w_k   = (const float*)d_in[7];
  const float* b_k   = (const float*)d_in[8];
  const float* w_v   = (const float*)d_in[9];
  const float* b_v   = (const float*)d_in[10];
  const float* w_pos = (const float*)d_in[11];
  const float* pbu   = (const float*)d_in[12];
  const float* pbv   = (const float*)d_in[13];
  const float* w_out = (const float*)d_in[14];
  const float* b_out = (const float*)d_in[15];
  float* out = (float*)d_out;

  const long NE = (long)B_ * T_ * E_;       // 2,097,152 elements
  _Float16* ws  = (_Float16*)d_ws;          // 6 x 4MB f16 buffers = 24MB
  _Float16* quB = ws + 0 * NE;
  _Float16* qvB = ws + 1 * NE;
  _Float16* kB  = ws + 2 * NE;
  _Float16* pB  = ws + 3 * NE;
  _Float16* vTB = ws + 4 * NE;
  _Float16* aoB = ws + 5 * NE;

  const dim3 gg(64, 8), bb(128);
  proj_gemm_kernel<<<gg, bb, 0, stream>>>(query, w_q, b_q, pbu, pbv, quB, qvB, 3);
  proj_gemm_kernel<<<gg, bb, 0, stream>>>(key,   w_k, b_k, nullptr, nullptr, kB,  nullptr, 0);
  proj_gemm_kernel<<<gg, bb, 0, stream>>>(value, w_v, b_v, nullptr, nullptr, vTB, nullptr, 1);
  proj_gemm_kernel<<<gg, bb, 0, stream>>>(pos,   w_pos, nullptr, nullptr, nullptr, pB, nullptr, 0);
  relattn_kernel<<<dim3(512), bb, 0, stream>>>(quB, qvB, kB, pB, vTB, mask, aoB);
  out_gemm_kernel<<<gg, bb, 0, stream>>>(aoB, w_out, b_out, out);
}
